// MSRPLoss_55851754717414
// MI455X (gfx1250) — compile-verified, matrix operand order checked by reading, not run on hardware
//
#include <hip/hip_runtime.h>

typedef __attribute__((ext_vector_type(16))) _Float16 v16h;
typedef __attribute__((ext_vector_type(8)))  float    v8f;

#define B_ROWS 16384
#define C_COLS 2048
#define NBLK   2048   // 8 waves/block * 2048 blocks = 16384 waves = one row per wave

// Kernel 1: per-row hinge margins, WMMA(ones) reduction, per-block partial sums.
__global__ __launch_bounds__(256) void hinge_partial_kernel(
    const float* __restrict__ outputs,
    const long long* __restrict__ labels,
    float* __restrict__ partial)
{
    const int tid  = threadIdx.x;
    const int lane = tid & 31;
    const int wave = tid >> 5;
    const int row  = blockIdx.x * 8 + wave;   // exact: NBLK*8 == B_ROWS

    // Ground-truth score for this row (uniform address -> broadcast load).
    const long long y = labels[row];
    const float g    = outputs[(size_t)row * C_COLS + (size_t)y];
    const float bias = 1.0f - g;

    // B matrix = all ones (f16): D[m,n] += sum_k A[m,k]
    v16h ones;
#pragma unroll
    for (int i = 0; i < 16; ++i) ones[i] = (_Float16)1.0f;

    v8f acc = {};

    const float4* rowp = (const float4*)(outputs + (size_t)row * C_COLS);

    // 2048 floats/row = 4 chunks of 512; each lane contributes 16 f16 values per WMMA.
#pragma unroll
    for (int chunk = 0; chunk < 4; ++chunk) {
        float4 x[4];
#pragma unroll
        for (int s = 0; s < 4; ++s)
            x[s] = rowp[chunk * 128 + s * 32 + lane];   // 32 lanes * 16B contiguous

        v16h a;
#pragma unroll
        for (int s = 0; s < 4; ++s) {
            a[4 * s + 0] = (_Float16)fmaxf(x[s].x + bias, 0.0f);
            a[4 * s + 1] = (_Float16)fmaxf(x[s].y + bias, 0.0f);
            a[4 * s + 2] = (_Float16)fmaxf(x[s].z + bias, 0.0f);
            a[4 * s + 3] = (_Float16)fmaxf(x[s].w + bias, 0.0f);
        }
        // D = A x ones + C : each of the 256 f32 cells accumulates a row-sum of A.
        acc = __builtin_amdgcn_wmma_f32_16x16x32_f16(
            /*neg_a=*/false, a, /*neg_b=*/false, ones,
            /*c_mod=*/(short)0, acc, /*reuse_a=*/false, /*reuse_b=*/false);
    }

    // Per-lane sum of 8 accumulator VGPRs; every row-sum is replicated 16x
    // across columns of D, so the block total gets scaled by 1/16 (exact).
    float s = acc[0] + acc[1] + acc[2] + acc[3] +
              acc[4] + acc[5] + acc[6] + acc[7];

    __shared__ float red[256];
    red[tid] = s;
    __syncthreads();
#pragma unroll
    for (int off = 128; off > 0; off >>= 1) {
        if (tid < off) red[tid] += red[tid + off];
        __syncthreads();
    }
    if (tid == 0) partial[blockIdx.x] = red[0] * 0.0625f;
}

// Kernel 2: deterministic final reduction of NBLK partials (double accum),
// subtract B_ROWS (the excluded ground-truth terms, each exactly 1.0).
__global__ __launch_bounds__(256) void hinge_final_kernel(
    const float* __restrict__ partial,
    float* __restrict__ out)
{
    const int tid = threadIdx.x;
    double s = 0.0;
#pragma unroll
    for (int i = tid; i < NBLK; i += 256) s += (double)partial[i];

    __shared__ double red[256];
    red[tid] = s;
    __syncthreads();
#pragma unroll
    for (int off = 128; off > 0; off >>= 1) {
        if (tid < off) red[tid] += red[tid + off];
        __syncthreads();
    }
    if (tid == 0) out[0] = (float)(red[0] - (double)B_ROWS);
}

extern "C" void kernel_launch(void* const* d_in, const int* in_sizes, int n_in,
                              void* d_out, int out_size, void* d_ws, size_t ws_size,
                              hipStream_t stream) {
    const float*     outputs = (const float*)d_in[0];      // [16384, 2048] f32
    const long long* labels  = (const long long*)d_in[1];  // [16384] i64
    float* partial = (float*)d_ws;                         // NBLK f32 partials
    float* outp    = (float*)d_out;                        // scalar f32

    hinge_partial_kernel<<<NBLK, 256, 0, stream>>>(outputs, labels, partial);
    hinge_final_kernel<<<1, 256, 0, stream>>>(partial, outp);
}
